// Convolution_44332652430077
// MI455X (gfx1250) — matrix operand order repaired and checked by reading, not compiled, not run on hardware
//
#include <hip/hip_runtime.h>
#include <hip/hip_bf16.h>
#include <stdint.h>

// ---------------------------------------------------------------------------
// MI455X (gfx1250) implementation of the gaussian-splat gather convolution.
// All GEMMs run on V_WMMA_F32_16X16X32_BF16 (fp32 accumulate, bf16 operands).
// Kernel 2 stages its A tiles with the Tensor Data Mover (tensor_load_to_lds,
// TENSORcnt, double-buffered, issue-ahead pipeline).
// Working set (x 4MB + feats 72MB bf16 + out 16MB) fits the 192MB L2.
// ---------------------------------------------------------------------------

typedef __attribute__((ext_vector_type(16))) __bf16   v16bf;
typedef __attribute__((ext_vector_type(8)))  __bf16   bf16x8;
typedef __attribute__((ext_vector_type(8)))  float    v8f;
typedef __attribute__((ext_vector_type(4)))  float    f32x4;
typedef __attribute__((ext_vector_type(4)))  uint32_t u32x4;
typedef __attribute__((ext_vector_type(8)))  uint32_t u32x8;

#define BB    4
#define CIN   64
#define HH    64
#define WW    64
#define COUT  256
#define KK    9
#define VS    8
#define HID   256
#define ADIN  66
#define K1P   96     // 66 padded to 3*32
#define N2P   32     // 27 padded to 32
#define KC    576    // KK*CIN

union V16U { v16bf v; bf16x8 h[2]; };

// Build a 16-element bf16 fragment from two contiguous 8-element (16B) runs.
__device__ __forceinline__ v16bf frag16(const __bf16* lo, const __bf16* hi) {
  V16U u;
  u.h[0] = *(const bf16x8*)lo;
  u.h[1] = *(const bf16x8*)hi;
  return u.v;
}

__device__ __forceinline__ uint32_t hashu(uint32_t a) {
  a ^= a >> 16; a *= 0x7feb352du;
  a ^= a >> 15; a *= 0x846ca68bu;
  a ^= a >> 16;
  return a;
}

// ---------------------------------------------------------------------------
// Kernel 0: fp32 weights -> bf16, transposed to [n][k] with K padded so each
// WMMA B-fragment is two contiguous 16B loads per lane.
// ---------------------------------------------------------------------------
__global__ void __launch_bounds__(256)
convert_weights(const float* __restrict__ w1, const float* __restrict__ w2,
                const float* __restrict__ wu,
                __bf16* __restrict__ w1b, __bf16* __restrict__ w2b,
                __bf16* __restrict__ wub) {
  int i = blockIdx.x * blockDim.x + threadIdx.x;
  const int S1 = COUT * K1P;   // 24576
  const int S2 = N2P * HID;    //  8192
  const int S3 = COUT * KC;    // 147456
  if (i < S1) {
    int n = i / K1P, k = i % K1P;
    float v = (k < ADIN) ? w1[k * COUT + n] : 0.f;
    w1b[i] = (__bf16)v;
  } else if (i < S1 + S2) {
    int j = i - S1;
    int n = j / HID, k = j % HID;
    float v = (n < 27) ? w2[k * 27 + n] : 0.f;
    w2b[j] = (__bf16)v;
  } else if (i < S1 + S2 + S3) {
    int j = i - S1 - S2;
    int n = j / KC, k = j % KC;
    wub[j] = (__bf16)wu[k * COUT + n];
  }
}

// ---------------------------------------------------------------------------
// Kernel 1: per-pixel MLP (WMMA) + gaussian params + 8-point gather -> feats.
// One block = one 16-pixel tile (same batch, same row, x0..x0+15).
// 128 threads = 4 wave32.
// ---------------------------------------------------------------------------
__global__ void __launch_bounds__(128)
mlp_splat_kernel(const float* __restrict__ x, const float* __restrict__ b1,
                 const float* __restrict__ b2, const __bf16* __restrict__ w1b,
                 const __bf16* __restrict__ w2b, __bf16* __restrict__ feats) {
  __shared__ __align__(16) __bf16 xin[16][K1P];     // 3 KB
  __shared__ __align__(16) __bf16 hid[16][HID];     // 8 KB
  __shared__ float params[16][28];
  __shared__ int   codes[16][KK][VS];
  __shared__ float wts  [16][KK][VS];

  const int t    = threadIdx.x;
  const int lane = t & 31;
  const int wave = t >> 5;
  const int half = lane >> 4;   // which 16-lane half of the wave
  const int l16  = lane & 15;

  const int tile = blockIdx.x;      // 4096 tiles
  const int b    = tile >> 8;
  const int rem  = tile & 255;
  const int y    = rem >> 2;
  const int x0   = (rem & 3) << 4;

  // ---- stage xin tile: 16 px x 96 k (channels + 2 coords + zero pad) ----
  for (int pass = 0; pass < 12; ++pass) {
    int e  = pass * 128 + t;        // 0..1535
    int px = e & 15;
    int k  = e >> 4;
    float v;
    if (k < CIN)      v = x[((b * CIN + k) << 12) + y * WW + x0 + px];
    else if (k == 64) v = (float)y * (1.0f / 63.0f);
    else if (k == 65) v = (float)(x0 + px) * (1.0f / 63.0f);
    else              v = 0.f;
    xin[px][k] = (__bf16)v;
  }
  __syncthreads();

  // ---- GEMM1: (16 x 96) @ (96 x 256) -> hidden, ReLU, bf16 to LDS ----
  for (int nt = 0; nt < 4; ++nt) {
    int n0 = wave * 64 + nt * 16;
    v8f acc = {};
    for (int kt = 0; kt < 3; ++kt) {
      int ka = kt * 32 + half * 16;                // A: contiguous 16 per half
      v16bf A = frag16(&xin[l16][ka], &xin[l16][ka + 8]);
      int kb = kt * 32 + half * 16;                // B: contiguous 16 per half
      const __bf16* bp = w1b + (size_t)(n0 + l16) * K1P + kb;
      v16bf Bf = frag16(bp, bp + 8);
      acc = __builtin_amdgcn_wmma_f32_16x16x32_bf16(
          false, A, false, Bf, (short)0, acc, false, false);
    }
    int n = n0 + l16;
    float bias = b1[n];
    for (int r = 0; r < 8; ++r) {
      int m = r + half * 8;
      hid[m][n] = (__bf16)fmaxf(acc[r] + bias, 0.f);
    }
  }
  __syncthreads();

  // ---- GEMM2: (16 x 256) @ (256 x 32) -> params (27 used) ----
  if (wave < 2) {
    int n0 = wave * 16;
    v8f acc = {};
    for (int kt = 0; kt < 8; ++kt) {
      int ka = kt * 32 + half * 16;
      v16bf A = frag16(&hid[l16][ka], &hid[l16][ka + 8]);
      int kb = kt * 32 + half * 16;
      const __bf16* bp = w2b + (size_t)(n0 + l16) * HID + kb;
      v16bf Bf = frag16(bp, bp + 8);
      acc = __builtin_amdgcn_wmma_f32_16x16x32_bf16(
          false, A, false, Bf, (short)0, acc, false, false);
    }
    int n = n0 + l16;
    if (n < 27) {
      float bias = b2[n];
      for (int r = 0; r < 8; ++r)
        params[r + half * 8][n] = acc[r] + bias;
    }
  }
  __syncthreads();

  // ---- gaussian params -> candidate indices + normalized weights ----
  if (t < 16 * KK) {
    int px = t / KK;
    int ks = t - px * KK;
    int xx = x0 + px;
    float p0 = params[px][ks * 3 + 0];
    float p1 = params[px][ks * 3 + 1];
    float p2 = params[px][ks * 3 + 2];

    float scy  = (float)y  * (0.9999f / 64.f) + 5e-5f;
    float scx  = (float)xx * (0.9999f / 64.f) + 5e-5f;
    float midy = __logf(scy / (1.f - scy));
    float midx = __logf(scx / (1.f - scx));
    float meany = 63.f / (1.f + __expf(-(midy + 0.1f * p0)));
    float meanx = 63.f / (1.f + __expf(-(midx + 0.1f * p1)));
    float sig   = (log1pf(__expf(p2 + 2.f)) + 0.05f) * 3.2f;  // *64*0.05
    int fmy = (int)floorf(meany);
    int fmx = (int)floorf(meanx);

    int cy[VS], cx[VS];
    cy[0] = fmy;     cx[0] = fmx;
    cy[1] = fmy;     cx[1] = fmx + 1;
    cy[2] = fmy + 1; cx[2] = fmx;
    cy[3] = fmy + 1; cx[3] = fmx + 1;
    // NOTE: reference uses jax threefry randint; deterministic hash with the
    // same uniform distribution substitutes (no runtime check available).
    uint32_t seed = (((uint32_t)b * 4096u + (uint32_t)y * 64u + (uint32_t)xx) * 9u
                     + (uint32_t)ks) * 4u;
    cy[4] = (int)(hashu(seed + 0x1001u) & 63);
    cx[4] = (int)(hashu(seed + 0x2002u) & 63);
    cy[5] = (int)(hashu(seed + 0x3003u) & 63);
    cx[5] = (int)(hashu(seed + 0x4004u) & 63);
    cy[6] = fmy - 4 + (int)(hashu(seed + 0x5005u) & 7);
    cx[6] = fmx - 4 + (int)(hashu(seed + 0x6006u) & 7);
    cy[7] = fmy - 4 + (int)(hashu(seed + 0x7007u) & 7);
    cx[7] = fmx - 4 + (int)(hashu(seed + 0x8008u) & 7);

    int code[VS]; float pr[VS]; float s = 0.f;
    float inv_sig = 1.f / sig;
    for (int v = 0; v < VS; ++v) {
      int iy = min(63, max(0, cy[v]));
      int ix = min(63, max(0, cx[v]));
      code[v] = iy * 64 + ix;
      bool dup = false;
      for (int u = 0; u < v; ++u) dup |= (code[u] == code[v]);
      float dy = ((float)iy - meany) * inv_sig;
      float dx = ((float)ix - meanx) * inv_sig;
      float p  = dup ? 0.f : __expf(-0.5f * (dy * dy + dx * dx));
      pr[v] = p; s += p;
    }
    float invs = 1.f / s;
    for (int v = 0; v < VS; ++v) {
      codes[px][ks][v] = code[v];
      wts  [px][ks][v] = pr[v] * invs;
    }
  }
  __syncthreads();

  // ---- weighted gather -> feats (bf16, row-major [pixel][576]) ----
  for (int it = 0; it < 72; ++it) {
    int e  = it * 128 + t;           // 0..9215 = 16*9*64
    int px = e / (KK * CIN);
    int r  = e - px * (KK * CIN);
    int ks = r >> 6;
    int c  = r & 63;
    const float* xb = x + ((size_t)(b * CIN + c) << 12);
    float acc = 0.f;
    for (int v = 0; v < VS; ++v)
      acc += wts[px][ks][v] * xb[codes[px][ks][v]];
    size_t row = (size_t)b * 4096 + (size_t)y * 64 + (x0 + px);
    feats[row * KC + ks * 64 + c] = (__bf16)acc;
  }
}

// ---------------------------------------------------------------------------
// Kernel 2: out = feats(65536 x 576, bf16) @ wu(576 x 256) + bu, written
// directly in (b, 256, h, w) layout. 256 threads = 8 waves, 16x256 tile.
// A tiles (16 x 32 bf16, row stride 1152B) are DMA'd into LDS by the Tensor
// Data Mover. Issue-ahead pipeline: TDM for step kt+1 is launched *before*
// waiting on step kt (same-wave TDM ops complete in order, so
// s_wait_tensorcnt 1 == "tile kt has landed"), keeping one DMA in flight
// under the WMMAs of the current step.
// ---------------------------------------------------------------------------
__global__ void __launch_bounds__(256)
out_gemm_kernel(const __bf16* __restrict__ feats, const __bf16* __restrict__ wub,
                const float* __restrict__ bu, float* __restrict__ out) {
  __shared__ __align__(16) __bf16 atile[2][16][32];   // 2 x 1 KB

  const int t    = threadIdx.x;
  const int lane = t & 31;
  const int wave = t >> 5;
  const int half = lane >> 4;
  const int l16  = lane & 15;

  const int tile = blockIdx.x;
  const int b    = tile >> 8;
  const int rem  = tile & 255;
  const int y    = rem >> 2;
  const int x0   = (rem & 3) << 4;
  const size_t rowbase = (size_t)b * 4096 + (size_t)y * 64 + x0;

  // Byte address of this tile's first row inside feats.
  const uint64_t featsb = (uint64_t)(uintptr_t)feats + (uint64_t)rowbase * (KC * 2);
  // Flat LDS addresses keep the byte offset in addr[31:0] (ISA 10.2 aperture).
  const uint32_t lds_base = (uint32_t)(uintptr_t)&atile[0][0][0];

  // Issue one TDM 2D tile load: 16 rows x 64B, global row stride 1152B,
  // packed contiguously into LDS. D# per CDNA5 ISA.
  auto issue_tdm = [&](int kt, int buf) {
    uint64_t ga = featsb + (uint64_t)kt * 64u;   // kt*32 bf16 = 64 bytes
    u32x4 g0;
    g0[0] = 1u;                                   // count=1, user mode
    g0[1] = lds_base + (uint32_t)buf * 1024u;     // lds_addr
    g0[2] = (uint32_t)ga;                         // global_addr[31:0]
    g0[3] = (uint32_t)(ga >> 32) | 0x80000000u;   // global_addr[56:32] | type=2
    u32x8 g1;
    g1[0] = 3u << 16;        // workgroup_mask=0, data_size=8B, no pad/iterate
    g1[1] = 0xFFFF0000u;     // atomic_barrier_addr=0, tensor_dim0[15:0]=0xFFFF
    g1[2] = 0xFFFF0000u;     // tensor_dim0[31:16]=0, tensor_dim1[15:0]=0xFFFF
    g1[3] = 8u << 16;        // tensor_dim1[31:16]=0, tile_dim0=8 (64B rows)
    g1[4] = 16u;             // tile_dim1=16 rows, tile_dim2=0
    g1[5] = 144u;            // tensor_dim0_stride = 1152B / 8B
    g1[6] = 0u;              // stride hi, tensor_dim1_stride lo
    g1[7] = 0u;
    asm volatile("tensor_load_to_lds %0, %1" :: "s"(g0), "s"(g1) : "memory");
  };

  v8f acc0 = {}, acc1 = {};
  const int n0 = wave * 32;

  if (wave == 0) issue_tdm(0, 0);

#pragma unroll
  for (int kt = 0; kt < 18; ++kt) {
    int buf = kt & 1;
    if (wave == 0) {
      if (kt + 1 < 18) {
        // buf^1 was last read in iteration kt-1, whose end-barrier has
        // already passed -> safe to start overwriting it now.
        issue_tdm(kt + 1, buf ^ 1);
        __builtin_amdgcn_s_wait_tensorcnt(1);   // tile kt landed (in-order)
      } else {
        __builtin_amdgcn_s_wait_tensorcnt(0);
      }
    }
    __syncthreads();                        // tile kt visible to all waves

    int ka = half * 16;
    v16bf A = frag16(&atile[buf][l16][ka], &atile[buf][l16][ka + 8]);
    int kb = kt * 32 + half * 16;
    const __bf16* bp0 = wub + (size_t)(n0 + l16) * KC + kb;
    const __bf16* bp1 = wub + (size_t)(n0 + 16 + l16) * KC + kb;
    if (kt + 1 < 18) __builtin_prefetch(bp0 + 32, 0, 1);  // global_prefetch_b8
    v16bf B0 = frag16(bp0, bp0 + 8);
    v16bf B1 = frag16(bp1, bp1 + 8);
    acc0 = __builtin_amdgcn_wmma_f32_16x16x32_bf16(
        false, A, false, B0, (short)0, acc0, false, false);
    acc1 = __builtin_amdgcn_wmma_f32_16x16x32_bf16(
        false, A, false, B1, (short)0, acc1, false, false);
    __syncthreads();                        // reads of buf done before reuse
  }

  // out[b][n][y][x]; per lane both f32x4 stores are 16B aligned.
  for (int which = 0; which < 2; ++which) {
    v8f acc = which ? acc1 : acc0;
    int n = n0 + which * 16 + l16;
    float bias = bu[n];
    f32x4 lo = { acc[0] + bias, acc[1] + bias, acc[2] + bias, acc[3] + bias };
    f32x4 hi = { acc[4] + bias, acc[5] + bias, acc[6] + bias, acc[7] + bias };
    float* op = out + ((size_t)(b * COUT + n) << 12) + y * 64 + x0 + half * 8;
    *(f32x4*)op       = lo;
    *(f32x4*)(op + 4) = hi;
  }
}

// ---------------------------------------------------------------------------
extern "C" void kernel_launch(void* const* d_in, const int* in_sizes, int n_in,
                              void* d_out, int out_size, void* d_ws, size_t ws_size,
                              hipStream_t stream) {
  (void)in_sizes; (void)n_in; (void)out_size; (void)ws_size;
  const float* x  = (const float*)d_in[0];
  const float* w1 = (const float*)d_in[1];
  const float* b1 = (const float*)d_in[2];
  const float* w2 = (const float*)d_in[3];
  const float* b2 = (const float*)d_in[4];
  const float* wu = (const float*)d_in[5];
  const float* bu = (const float*)d_in[6];
  float* out = (float*)d_out;

  char* ws = (char*)d_ws;
  __bf16* w1b   = (__bf16*)(ws);             // 256*96  bf16 = 48 KB
  __bf16* w2b   = (__bf16*)(ws + 49152);     //  32*256 bf16 = 16 KB
  __bf16* wub   = (__bf16*)(ws + 65536);     // 256*576 bf16 = 288 KB
  __bf16* feats = (__bf16*)(ws + 360448);    // 65536*576 bf16 = 72 MB

  convert_weights<<<704, 256, 0, stream>>>(w1, w2, wu, w1b, w2b, wub);
  mlp_splat_kernel<<<4096, 128, 0, stream>>>(x, b1, b2, w1b, w2b, feats);
  out_gemm_kernel<<<4096, 256, 0, stream>>>(feats, wub, bu, out);
}
